// KernalAttention_52304111730698
// MI455X (gfx1250) — matrix operand order repaired and checked by reading, not compiled, not run on hardware
//
#include <hip/hip_runtime.h>

// Problem constants (match reference)
constexpr int kB  = 4;
constexpr int kN  = 4096;
constexpr int kH  = 8;
constexpr int kD  = 64;
constexpr int kR  = 256;
constexpr int kNH = kN * kH;        // 32768 rows per batch (flattened n*H+h)
constexpr int kSLICES = 16;         // split-K over N in phase 1
constexpr float kLOG2E = 1.44269504088896340736f;

typedef __attribute__((ext_vector_type(16))) __bf16          v16bf;
typedef __attribute__((ext_vector_type(2)))  __bf16          v2bf;
typedef __attribute__((ext_vector_type(8)))  float           v8f;
typedef __attribute__((ext_vector_type(8)))  unsigned short  v8us;

union BFrag {
  v16bf v;
  unsigned short u[16];
  unsigned w[8];          // w[i] = {u[2i+1], u[2i]}
  v8us h[2];
};

// Pack two f32 -> one dword of two bf16. Prefer the dedicated builtin; else use
// language-level fptrunc (backend selects the native cvt when available).
__device__ __forceinline__ unsigned pk2bf(float a, float b) {
#if __has_builtin(__builtin_amdgcn_cvt_pk_bf16_f32)
  return __builtin_bit_cast(unsigned, __builtin_amdgcn_cvt_pk_bf16_f32(a, b));
#else
  v2bf p = { (__bf16)a, (__bf16)b };
  return __builtin_bit_cast(unsigned, p);
#endif
}

__device__ __forceinline__ unsigned short f2bf(float f) {
  return __builtin_bit_cast(unsigned short, (__bf16)f);
}

// ---------------------------------------------------------------------------
// Kernel 1: pack W [D,R] f32 -> bf16 B-fragment layout.
// Fragment (kk = d-ktile of 32, rr = r-tile of 16): per lane l, element j:
//   B[K = kk*32 + (l<16?0:8) + (j<8?j:j+8), N = rr*16 + (l&15)]
// stored contiguously: Wp[((kk*16+rr)*32 + l)*16 + j]
// ---------------------------------------------------------------------------
__global__ void pack_w_kernel(const float* __restrict__ W,
                              unsigned short* __restrict__ Wp) {
  int t = blockIdx.x * blockDim.x + threadIdx.x;   // 16384 threads
  int j  = t & 15;
  int l  = (t >> 4) & 31;
  int rr = (t >> 9) & 15;
  int kk = t >> 13;
  int d = kk * 32 + ((l & 16) ? 8 : 0) + (j < 8 ? j : j + 8);
  int r = rr * 16 + (l & 15);
  Wp[t] = f2bf(W[d * kR + r]);
}

// ---------------------------------------------------------------------------
// Kernel 2: kv partials. One wave per (b, rtile, slice).
//   S[n, r]  = K[b,n,:] @ W[:,r]   (2 bf16 WMMAs per 16-n group)
//   wk       = exp(S)              (lane-local repack D-tile -> A-fragment)
//   kv[r, d] += wkT @ V            (4 bf16 WMMAs per 32-n chunk)
// ---------------------------------------------------------------------------
__global__ __launch_bounds__(128)
void phase1_kernel(const float* __restrict__ K, const float* __restrict__ V,
                   const unsigned short* __restrict__ Wp,
                   float* __restrict__ kvp /* [16][B][R][D] f32 */) {
  const int lane  = threadIdx.x & 31;
  const int wave  = threadIdx.x >> 5;
  const int wid   = blockIdx.x * 4 + wave;     // 1024 waves
  const int slice = wid & 15;
  const int rtile = (wid >> 4) & 15;
  const int b     = wid >> 8;
  const int r0    = rtile * 16;
  const int lm    = lane & 15;
  const int koff  = (lane & 16) ? 8 : 0;       // A-fragment K offset
  const int noff  = (lane & 16) ? 16 : 0;      // unused; B-fragments also use K offset 8
  const int rofs  = (lane & 16) ? 8 : 0;       // C/D-tile M offset

  // W B-fragments for this r-tile (both d-ktiles), contiguous 32B per lane
  BFrag wf0, wf1;
  {
    const v8us* p0 = (const v8us*)(Wp + ((size_t)(0 * 16 + rtile) * 32 + lane) * 16);
    const v8us* p1 = (const v8us*)(Wp + ((size_t)(1 * 16 + rtile) * 32 + lane) * 16);
    wf0.h[0] = p0[0]; wf0.h[1] = p0[1];
    wf1.h[0] = p1[0]; wf1.h[1] = p1[1];
  }

  v8f acc[4] = {};
  const float* Kb = K + (size_t)b * kN * kD;
  const float* Vb = V + (size_t)b * kN * kD;

  const int nbeg = slice * (kN / kSLICES);
  const int nend = nbeg + (kN / kSLICES);
  for (int n0 = nbeg; n0 < nend; n0 += 32) {
    // A fragments of K (rows n0..n0+15 and n0+16..n0+31, two d-ktiles each)
    BFrag a10, a11, a20, a21;
    const float* row1 = Kb + (size_t)(n0 + lm) * kD;
    const float* row2 = Kb + (size_t)(n0 + 16 + lm) * kD;
#pragma unroll
    for (int p = 0; p < 4; ++p) {              // pair index: elements 2p,2p+1
      const int j = 2 * p;
      a10.w[p]     = pk2bf(row1[koff + j],           row1[koff + j + 1]);
      a10.w[4 + p] = pk2bf(row1[koff + 16 + j],      row1[koff + 17 + j]);
      a11.w[p]     = pk2bf(row1[32 + koff + j],      row1[32 + koff + j + 1]);
      a11.w[4 + p] = pk2bf(row1[32 + koff + 16 + j], row1[32 + koff + 17 + j]);
      a20.w[p]     = pk2bf(row2[koff + j],           row2[koff + j + 1]);
      a20.w[4 + p] = pk2bf(row2[koff + 16 + j],      row2[koff + 17 + j]);
      a21.w[p]     = pk2bf(row2[32 + koff + j],      row2[32 + koff + j + 1]);
      a21.w[4 + p] = pk2bf(row2[32 + koff + 16 + j], row2[32 + koff + 17 + j]);
    }

    v8f s1 = {}, s2 = {};
    s1 = __builtin_amdgcn_wmma_f32_16x16x32_bf16(false, a10.v, false, wf0.v, (short)0, s1, false, false);
    s1 = __builtin_amdgcn_wmma_f32_16x16x32_bf16(false, a11.v, false, wf1.v, (short)0, s1, false, false);
    s2 = __builtin_amdgcn_wmma_f32_16x16x32_bf16(false, a20.v, false, wf0.v, (short)0, s2, false, false);
    s2 = __builtin_amdgcn_wmma_f32_16x16x32_bf16(false, a21.v, false, wf1.v, (short)0, s2, false, false);

    // exp(S) D-tile -> A-fragment of wk^T is LANE-LOCAL:
    //   a.u[j]   = wk[n0 + koff + j,      r0 + lm] = exp(s1[j])
    //   a.u[8+j] = wk[n0 + 16 + koff + j, r0 + lm] = exp(s2[j])
    BFrag awk;
#pragma unroll
    for (int p = 0; p < 4; ++p) {
      float e0 = __builtin_amdgcn_exp2f(s1[2 * p]     * kLOG2E);
      float e1 = __builtin_amdgcn_exp2f(s1[2 * p + 1] * kLOG2E);
      float e2 = __builtin_amdgcn_exp2f(s2[2 * p]     * kLOG2E);
      float e3 = __builtin_amdgcn_exp2f(s2[2 * p + 1] * kLOG2E);
      awk.w[p]     = pk2bf(e0, e1);
      awk.w[4 + p] = pk2bf(e2, e3);
    }

#pragma unroll
    for (int cc = 0; cc < 4; ++cc) {
      BFrag bv;  // B-fragment of V: K = n, N = d
      const float* vcol = Vb + (size_t)(n0 + koff) * kD + cc * 16 + lm;
#pragma unroll
      for (int p = 0; p < 8; ++p)
        bv.w[p] = pk2bf(vcol[(size_t)(2 * p + ((p & 4) ? 8 : 0)) * kD], vcol[(size_t)(2 * p + 1 + ((p & 4) ? 8 : 0)) * kD]);
      acc[cc] = __builtin_amdgcn_wmma_f32_16x16x32_bf16(false, awk.v, false, bv.v, (short)0, acc[cc], false, false);
    }
  }

  // store deterministic split-K partials
  float* dst = kvp + (((size_t)slice * kB + b) * kR + r0) * kD;
#pragma unroll
  for (int cc = 0; cc < 4; ++cc)
#pragma unroll
    for (int i = 0; i < 8; ++i)
      dst[(size_t)(i + rofs) * kD + cc * 16 + lm] = acc[cc][i];
}

// ---------------------------------------------------------------------------
// Kernel 3: reduce the 16 split-K slices and pack kv into bf16 B-fragment
// layout for phase 2:  kvpk[(((b*8+kk)*4+cc)*32 + l)*16 + j]
//   = bf16( kv[b, kk*32 + (l<16?0:8) + (j<8?j:j+8), cc*16 + (l&15)] )
// ---------------------------------------------------------------------------
__global__ void pack_kv_kernel(const float* __restrict__ kvp,
                               unsigned short* __restrict__ kvpk) {
  int t = blockIdx.x * blockDim.x + threadIdx.x;   // 65536 threads
  int j  = t & 15;
  int l  = (t >> 4) & 31;
  int cc = (t >> 9) & 3;
  int kk = (t >> 11) & 7;
  int b  = t >> 14;
  int r = kk * 32 + ((l & 16) ? 8 : 0) + (j < 8 ? j : j + 8);
  int d = cc * 16 + (l & 15);
  size_t idx = ((size_t)b * kR + r) * kD + d;
  float s = 0.f;
#pragma unroll
  for (int sl = 0; sl < kSLICES; ++sl)
    s += kvp[(size_t)sl * (kB * kR * kD) + idx];
  kvpk[t] = f2bf(s);
}

// ---------------------------------------------------------------------------
// Kernel 4 (dominant): one wave per 16-row Q tile.
//   S  = Q_tile @ W        (32 bf16 WMMAs)
//   wk = exp(S) -> LDS (bf16, row-major per wave; transposes lane roles)
//   out_tile = wk @ kv     (32 bf16 WMMAs, A-fragments via ds_load_b128)
// ---------------------------------------------------------------------------
__global__ __launch_bounds__(256)
void phase2_kernel(const float* __restrict__ Q,
                   const unsigned short* __restrict__ Wp,
                   const unsigned short* __restrict__ kvpk,
                   float* __restrict__ out) {
  __shared__ unsigned short lds[8 * 16 * 256];   // 8 waves x 8KB = 64KB
  const int lane = threadIdx.x & 31;
  const int wave = threadIdx.x >> 5;
  const int tile = blockIdx.x * 8 + wave;        // 8192 tiles
  const int b    = tile >> 11;                   // / 2048
  const int m0   = (tile & 2047) * 16;           // row-tile base in [NH]
  const int lm   = lane & 15;
  const int koff = (lane & 16) ? 8 : 0;
  const int rofs = (lane & 16) ? 8 : 0;
  unsigned short* wk = lds + wave * (16 * 256);

  // A-fragments of Q rows (two d-ktiles)
  BFrag aq0, aq1;
  const float* qrow = Q + ((size_t)b * kNH + m0 + lm) * kD;
#pragma unroll
  for (int p = 0; p < 4; ++p) {
    const int j = 2 * p;
    aq0.w[p]     = pk2bf(qrow[koff + j],           qrow[koff + j + 1]);
    aq0.w[4 + p] = pk2bf(qrow[koff + 16 + j],      qrow[koff + 17 + j]);
    aq1.w[p]     = pk2bf(qrow[32 + koff + j],      qrow[32 + koff + j + 1]);
    aq1.w[4 + p] = pk2bf(qrow[32 + koff + 16 + j], qrow[32 + koff + 17 + j]);
  }

  // GEMM1: S = Q @ W, exp -> bf16 into per-wave LDS (row-major [16][256])
#pragma unroll
  for (int rr = 0; rr < 16; ++rr) {
    BFrag wf0, wf1;
    const v8us* p0 = (const v8us*)(Wp + ((size_t)rr * 32 + lane) * 16);
    const v8us* p1 = (const v8us*)(Wp + ((size_t)(16 + rr) * 32 + lane) * 16);
    wf0.h[0] = p0[0]; wf0.h[1] = p0[1];
    wf1.h[0] = p1[0]; wf1.h[1] = p1[1];
    v8f s = {};
    s = __builtin_amdgcn_wmma_f32_16x16x32_bf16(false, aq0.v, false, wf0.v, (short)0, s, false, false);
    s = __builtin_amdgcn_wmma_f32_16x16x32_bf16(false, aq1.v, false, wf1.v, (short)0, s, false, false);
#pragma unroll
    for (int i = 0; i < 8; ++i)
      wk[(i + rofs) * 256 + rr * 16 + lm] =
          f2bf(__builtin_amdgcn_exp2f(s[i] * kLOG2E));
  }
  // same-wave LDS RAW: compiler inserts s_wait_dscnt; no barrier needed
  // (each wave touches only its own 8KB region).

  // GEMM2: out = wk @ kv
  v8f acc[4] = {};
#pragma unroll
  for (int kk = 0; kk < 8; ++kk) {
    BFrag awk;  // A-fragment: M = lm (row), K = kk*32 + koff + {0..7, 16..23}
    awk.h[0] = *(const v8us*)(wk + lm * 256 + kk * 32 + koff);
    awk.h[1] = *(const v8us*)(wk + lm * 256 + kk * 32 + koff + 16);
#pragma unroll
    for (int cc = 0; cc < 4; ++cc) {
      BFrag bkv;
      const v8us* bp =
          (const v8us*)(kvpk + (((size_t)(b * 8 + kk) * 4 + cc) * 32 + lane) * 16);
      bkv.h[0] = bp[0]; bkv.h[1] = bp[1];
      acc[cc] = __builtin_amdgcn_wmma_f32_16x16x32_bf16(false, awk.v, false, bkv.v, (short)0, acc[cc], false, false);
    }
  }

  float* orow = out + ((size_t)b * kNH + m0) * kD;
#pragma unroll
  for (int cc = 0; cc < 4; ++cc)
#pragma unroll
    for (int i = 0; i < 8; ++i)
      orow[(size_t)(i + rofs) * kD + cc * 16 + lm] = acc[cc][i];
}

// ---------------------------------------------------------------------------
extern "C" void kernel_launch(void* const* d_in, const int* in_sizes, int n_in,
                              void* d_out, int out_size, void* d_ws, size_t ws_size,
                              hipStream_t stream) {
  (void)in_sizes; (void)n_in; (void)out_size; (void)ws_size;
  const float* Q = (const float*)d_in[0];
  const float* K = (const float*)d_in[1];
  const float* V = (const float*)d_in[2];
  const float* W = (const float*)d_in[3];
  float* out = (float*)d_out;

  char* ws = (char*)d_ws;
  unsigned short* Wp   = (unsigned short*)ws;                       // 32 KB
  float*          kvp  = (float*)(ws + 32768);                      // 16*4*256*64*4 = 4 MB
  unsigned short* kvpk = (unsigned short*)(ws + 32768 +
                          (size_t)kSLICES * kB * kR * kD * sizeof(float)); // 128 KB

  pack_w_kernel <<<  64, 256, 0, stream>>>(W, Wp);
  phase1_kernel <<< 256, 128, 0, stream>>>(K, V, Wp, kvp);
  pack_kv_kernel<<< 256, 256, 0, stream>>>(kvp, kvpk);
  phase2_kernel <<<1024, 256, 0, stream>>>(Q, Wp, kvpk, out);
}
